// Model_29162827940742
// MI455X (gfx1250) — matrix operand order repaired and checked by reading, not compiled, not run on hardware
//
#include <hip/hip_runtime.h>

#define NNODES 6144
#define KNBR   32
#define NLAYER 6
#define DDIM   128
#define MDIM   64
#define HEDIM  256
#define NFEAT  68
#define EMBD   128
#define NPB    3        // nodes per edge-kernel workgroup (96 edge rows)

typedef __attribute__((ext_vector_type(16))) __bf16        v16bf;
typedef __attribute__((ext_vector_type(8)))  float         v8f;
typedef __attribute__((ext_vector_type(4)))  unsigned int  u32x4;

union Frag32 { u32x4 q[2]; v16bf v; };

__device__ __forceinline__ unsigned short f2bf(float f) {
  unsigned int u = __float_as_uint(f);
  u += 0x7FFFu + ((u >> 16) & 1u);          // round-to-nearest-even
  return (unsigned short)(u >> 16);
}

__device__ __forceinline__ float siluf(float x) {
  return x * (1.0f / (1.0f + __expf(-x)));
}

// A fragment (16x32 bf16): lane holds row (row0 + lane&15); K elements j=0..7 are
// [k0+8g, k0+8g+8) and j=8..15 are [k0+16+8g, +8), g = lane>>4.  Two 16B LDS loads.
__device__ __forceinline__ v16bf load_a_frag(const unsigned short* S, int row0, int stride, int k0) {
  int lane = threadIdx.x & 31;
  const unsigned short* p = S + (row0 + (lane & 15)) * stride + k0 + 8 * (lane >> 4);
  Frag32 f;
  f.q[0] = *(const u32x4*)(p);
  f.q[1] = *(const u32x4*)(p + 16);
  return f.v;
}

// B fragment: packed by pack_b_kernel so each lane reads 16 contiguous halves.
__device__ __forceinline__ v16bf load_b_frag(const unsigned short* P, int tile) {
  int lane = threadIdx.x & 31;
  const unsigned short* p = P + (size_t)tile * 512 + lane * 16;
  Frag32 f;
  f.q[0] = *(const u32x4*)(p);
  f.q[1] = *(const u32x4*)(p + 8);
  return f.v;
}

#define WMMA_BF16(a, b, c) \
  __builtin_amdgcn_wmma_f32_16x16x32_bf16(false, (a), false, (b), (short)0, (c), false, false)

// ---------------------------------------------------------------------------
// Repack f32 weight [K,N] (row-major) into per-tile B-fragment layout (bf16).
// B 32x16 layout: lane -> column n = nt*16 + (lane&15); K = kt*32 + 8*(lane>>4) + j (j<8), kt*32 + 16 + 8*(lane>>4) + (j-8) (j>=8).
// ---------------------------------------------------------------------------
__global__ void pack_b_kernel(const float* __restrict__ src, unsigned short* __restrict__ dst,
                              int Ksrc, int Nmat) {
  int NT = Nmat >> 4;
  int tile = blockIdx.x, lane = threadIdx.x;
  int kt = tile / NT, nt = tile % NT;
  int n = nt * 16 + (lane & 15);
  int kbase = kt * 32 + 8 * (lane >> 4);
  unsigned short* d = dst + (size_t)tile * 512 + lane * 16;
#pragma unroll
  for (int j = 0; j < 16; ++j) {
    int k = kbase + (j < 8 ? j : j + 8);
    float v = (k < Ksrc) ? src[(size_t)k * Nmat + n] : 0.0f;
    d[j] = f2bf(v);
  }
}

__global__ void zero_kernel(float* p) { p[threadIdx.x] = 0.0f; }

// ---------------------------------------------------------------------------
// Encoder: feats = x @ w_enc + b_enc   [6144,68]@[68,128], K padded to 96.
// ---------------------------------------------------------------------------
__global__ void encoder_kernel(const float* __restrict__ x, const unsigned short* __restrict__ Penc,
                               const float* __restrict__ b_enc,
                               float* __restrict__ featsF, unsigned short* __restrict__ featsH) {
  const int STR = 104;
  __shared__ u32x4 raw[832];                       // 64 * 104 halves
  unsigned short* S = (unsigned short*)raw;
  int r0 = blockIdx.x * 64, t = threadIdx.x, lane = t & 31, w = t >> 5, g = lane >> 4;
  for (int f = t; f < 64 * 96; f += 256) {
    int row = f / 96, col = f % 96;
    float v = (col < NFEAT) ? x[(size_t)(r0 + row) * NFEAT + col] : 0.0f;
    S[row * STR + col] = f2bf(v);
  }
  __syncthreads();
  int mt = w & 3, nh = w >> 2;
  v8f acc[4] = {};
  for (int kt = 0; kt < 3; ++kt) {
    v16bf a = load_a_frag(S, mt * 16, STR, kt * 32);
#pragma unroll
    for (int nt = 0; nt < 4; ++nt) {
      v16bf b = load_b_frag(Penc, kt * 8 + nh * 4 + nt);
      acc[nt] = WMMA_BF16(a, b, acc[nt]);
    }
  }
#pragma unroll
  for (int nt = 0; nt < 4; ++nt) {
    int col = (nh * 4 + nt) * 16 + (lane & 15);
#pragma unroll
    for (int j = 0; j < 8; ++j) {
      int row = r0 + mt * 16 + j + 8 * g;
      float v = acc[nt][j] + b_enc[col];
      featsF[(size_t)row * DDIM + col] = v;
      featsH[(size_t)row * DDIM + col] = f2bf(v);
    }
  }
}

// ---------------------------------------------------------------------------
// Neighbor structure.  ranking = -1 (self), 0 (adjacency), rel_dist (>0) else;
// top_k(-ranking, 32) == [self] ++ 31 smallest-column adjacency entries.
// ---------------------------------------------------------------------------
__global__ void nbr_kernel(const int* __restrict__ ei, const float* __restrict__ coords,
                           int* __restrict__ idxW, float* __restrict__ rdW) {
  __shared__ int sidx[32];
  int i = blockIdx.x, lane = threadIdx.x;
  int d = ei[(size_t)NNODES * KNBR + (size_t)i * KNBR + lane];
  int rank = 0;
  for (int j = 0; j < KNBR; ++j) {
    int dj = __shfl(d, j, 32);
    rank += (dj < d) ? 1 : 0;
  }
  if (rank < KNBR - 1) sidx[1 + rank] = d;       // keep 31 smallest columns
  if (lane == 0) sidx[0] = i;                    // self first (ranking -1)
  __syncthreads();
  int nb = sidx[lane];
  float dx = coords[3 * i + 0] - coords[3 * nb + 0];
  float dy = coords[3 * i + 1] - coords[3 * nb + 1];
  float dz = coords[3 * i + 2] - coords[3 * nb + 2];
  idxW[(size_t)i * KNBR + lane] = nb;
  rdW[(size_t)i * KNBR + lane] = dx * dx + dy * dy + dz * dz;
}

// ---------------------------------------------------------------------------
// Edge MLP + aggregation: 3 nodes (96 edge rows) per workgroup, 8 waves.
// GEMM1: [96,256]@[256,256] — wave owns 2 n-tiles x all 6 m-tiles, so each
//        packed-B tile is fetched ONCE per block (3x less L2 weight traffic
//        than one-node blocks).  Epilogue folds rank-1 rd*ew1[256,:] + bias.
// GEMM2: [96,256]@[256,64], silu, then per-node column sums via LDS.
// ---------------------------------------------------------------------------
__global__ void edge_kernel(const unsigned short* __restrict__ featsH,
                            const int* __restrict__ idxW, const float* __restrict__ rdW,
                            const unsigned short* __restrict__ Pew1,
                            const float* __restrict__ ew1last, const float* __restrict__ eb1,
                            const unsigned short* __restrict__ Pew2,
                            const float* __restrict__ eb2, float* __restrict__ m_i) {
  const int STR = 264;
  const int ROWS = 32 * NPB;                     // 96
  __shared__ u32x4 raw[ROWS * STR * 2 / 16];     // 50688 B; A-stage -> G1 -> m_ij
  __shared__ int   sidx[ROWS];
  __shared__ float srd[ROWS];
  unsigned short* S  = (unsigned short*)raw;
  float*          Sm = (float*)raw;              // m_ij f32 view (24576 B)
  int n0 = blockIdx.x * NPB;
  int t = threadIdx.x, lane = t & 31, w = t >> 5, g = lane >> 4;

  if (t < ROWS) {
    sidx[t] = idxW[(size_t)n0 * KNBR + t];       // contiguous across the 3 nodes
    srd[t]  = rdW[(size_t)n0 * KNBR + t];
  }
  __syncthreads();
  {                                              // stage [fi | fj], 4 threads/row
    int part = t & 3;
    for (int rr = t >> 2; rr < ROWS; rr += 64) {
      int p = rr >> 5;
      const u32x4* fj = (const u32x4*)(featsH + (size_t)sidx[rr] * DDIM + part * 32);
      const u32x4* fi = (const u32x4*)(featsH + (size_t)(n0 + p) * DDIM + part * 32);
      u32x4* dj = (u32x4*)(S + rr * STR + 128 + part * 32);
      u32x4* di = (u32x4*)(S + rr * STR + part * 32);
      dj[0] = fj[0]; dj[1] = fj[1]; dj[2] = fj[2]; dj[3] = fj[3];
      di[0] = fi[0]; di[1] = fi[1]; di[2] = fi[2]; di[3] = fi[3];
    }
  }
  __syncthreads();

  v8f acc[6][2] = {};
  for (int kt = 0; kt < 8; ++kt) {
    v16bf b0 = load_b_frag(Pew1, kt * 16 + 2 * w);
    v16bf b1 = load_b_frag(Pew1, kt * 16 + 2 * w + 1);
#pragma unroll
    for (int mt = 0; mt < 6; ++mt) {
      v16bf a = load_a_frag(S, mt * 16, STR, kt * 32);
      acc[mt][0] = WMMA_BF16(a, b0, acc[mt][0]);
      acc[mt][1] = WMMA_BF16(a, b1, acc[mt][1]);
    }
  }
  __syncthreads();                               // all A reads done before overwrite
#pragma unroll
  for (int mt = 0; mt < 6; ++mt)
#pragma unroll
    for (int nt = 0; nt < 2; ++nt) {
      int col = (2 * w + nt) * 16 + (lane & 15);
      float wl = ew1last[col], bb = eb1[col];
#pragma unroll
      for (int j = 0; j < 8; ++j) {
        int m = mt * 16 + j + 8 * g;
        float v = acc[mt][nt][j] + srd[m] * wl + bb;   // fold the 257th K row
        S[m * STR + col] = f2bf(siluf(v));
      }
    }
  __syncthreads();

  // GEMM2: wave w -> n-tile (w&3), m-tiles (w>>2)*3 .. +2
  int nt2 = w & 3, h = w >> 2;
  v8f acc2[3] = {};
  for (int kt = 0; kt < 8; ++kt) {
    v16bf b = load_b_frag(Pew2, kt * 4 + nt2);
#pragma unroll
    for (int q = 0; q < 3; ++q) {
      v16bf a = load_a_frag(S, (h * 3 + q) * 16, STR, kt * 32);
      acc2[q] = WMMA_BF16(a, b, acc2[q]);
    }
  }
  __syncthreads();                               // done reading S; reuse as m_ij f32
  int col2 = nt2 * 16 + (lane & 15);
  float bb2 = eb2[col2];
#pragma unroll
  for (int q = 0; q < 3; ++q)
#pragma unroll
    for (int j = 0; j < 8; ++j) {
      int row = (h * 3 + q) * 16 + j + 8 * g;
      Sm[row * MDIM + col2] = siluf(acc2[q][j] + bb2);
    }
  __syncthreads();
  if (t < NPB * MDIM) {                          // per-node column sums over 32 edges
    int p = t >> 6, col = t & 63;
    float s = 0.0f;
#pragma unroll
    for (int r = 0; r < 32; ++r) s += Sm[(p * 32 + r) * MDIM + col];
    m_i[(size_t)(n0 + p) * MDIM + col] = s;
  }
}

// ---------------------------------------------------------------------------
// Node MLP + residual, 64 rows / block, 8 waves.
// ---------------------------------------------------------------------------
__global__ void node_kernel(float* __restrict__ featsF, unsigned short* __restrict__ featsH,
                            const float* __restrict__ m_i,
                            const unsigned short* __restrict__ Pnw1, const float* __restrict__ nb1,
                            const unsigned short* __restrict__ Pnw2, const float* __restrict__ nb2) {
  const int HSTR = 200, GSTR = 264;
  __shared__ u32x4 raw[3712];                    // 64*200 + 64*264 halves = 59392 B
  unsigned short* H = (unsigned short*)raw;
  unsigned short* G = H + 64 * HSTR;
  int r0 = blockIdx.x * 64, t = threadIdx.x, lane = t & 31, w = t >> 5, g = lane >> 4;

  for (int f = t; f < 64 * HSTR; f += 256) {
    int row = f / HSTR, col = f % HSTR;
    float v = 0.0f;
    if (col < 128)      v = featsF[(size_t)(r0 + row) * DDIM + col];
    else if (col < 192) v = m_i[(size_t)(r0 + row) * MDIM + (col - 128)];
    H[f] = f2bf(v);
  }
  __syncthreads();

  int mt = w & 3, nh = w >> 2;
  v8f acc[8] = {};
  for (int kt = 0; kt < 6; ++kt) {
    v16bf a = load_a_frag(H, mt * 16, HSTR, kt * 32);
#pragma unroll
    for (int nt = 0; nt < 8; ++nt) {
      v16bf b = load_b_frag(Pnw1, kt * 16 + nh * 8 + nt);
      acc[nt] = WMMA_BF16(a, b, acc[nt]);
    }
  }
#pragma unroll
  for (int nt = 0; nt < 8; ++nt) {
    int col = (nh * 8 + nt) * 16 + (lane & 15);
#pragma unroll
    for (int j = 0; j < 8; ++j) {
      int m = mt * 16 + j + 8 * g;
      G[m * GSTR + col] = f2bf(siluf(acc[nt][j] + nb1[col]));
    }
  }
  __syncthreads();

  v8f acc2[4] = {};
  for (int kt = 0; kt < 8; ++kt) {
    v16bf a = load_a_frag(G, mt * 16, GSTR, kt * 32);
#pragma unroll
    for (int nt = 0; nt < 4; ++nt) {
      v16bf b = load_b_frag(Pnw2, kt * 8 + nh * 4 + nt);
      acc2[nt] = WMMA_BF16(a, b, acc2[nt]);
    }
  }
#pragma unroll
  for (int nt = 0; nt < 4; ++nt) {
    int col = (nh * 4 + nt) * 16 + (lane & 15);
#pragma unroll
    for (int j = 0; j < 8; ++j) {
      int row = r0 + mt * 16 + j + 8 * g;
      float v = acc2[nt][j] + nb2[col] + featsF[(size_t)row * DDIM + col];
      featsF[(size_t)row * DDIM + col] = v;
      featsH[(size_t)row * DDIM + col] = f2bf(v);
    }
  }
}

// ---------------------------------------------------------------------------
// node_dec: silu(feats@dw1+db1)@dw2+db2, column-summed into gsum[128].
// ---------------------------------------------------------------------------
__global__ void dec_kernel(const float* __restrict__ featsF,
                           const unsigned short* __restrict__ Pdw1, const float* __restrict__ db1,
                           const unsigned short* __restrict__ Pdw2, const float* __restrict__ db2,
                           float* __restrict__ gsum) {
  const int STR = 136;
  __shared__ u32x4 raw[2176];                    // 2 * 64*136 halves
  __shared__ float gpart[128];
  unsigned short* H = (unsigned short*)raw;
  unsigned short* G = H + 64 * STR;
  int r0 = blockIdx.x * 64, t = threadIdx.x, lane = t & 31, w = t >> 5, g = lane >> 4;

  if (t < 128) gpart[t] = 0.0f;
  for (int f = t; f < 64 * STR; f += 256) {
    int col = f % STR;
    int row = f / STR;
    float v = (col < 128) ? featsF[(size_t)(r0 + row) * DDIM + col] : 0.0f;
    H[f] = f2bf(v);
  }
  __syncthreads();

  int mt = w & 3, nh = w >> 2;
  v8f acc[4] = {};
  for (int kt = 0; kt < 4; ++kt) {
    v16bf a = load_a_frag(H, mt * 16, STR, kt * 32);
#pragma unroll
    for (int nt = 0; nt < 4; ++nt) {
      v16bf b = load_b_frag(Pdw1, kt * 8 + nh * 4 + nt);
      acc[nt] = WMMA_BF16(a, b, acc[nt]);
    }
  }
#pragma unroll
  for (int nt = 0; nt < 4; ++nt) {
    int col = (nh * 4 + nt) * 16 + (lane & 15);
#pragma unroll
    for (int j = 0; j < 8; ++j) {
      int m = mt * 16 + j + 8 * g;
      G[m * STR + col] = f2bf(siluf(acc[nt][j] + db1[col]));
    }
  }
  __syncthreads();

  v8f acc2[4] = {};
  for (int kt = 0; kt < 4; ++kt) {
    v16bf a = load_a_frag(G, mt * 16, STR, kt * 32);
#pragma unroll
    for (int nt = 0; nt < 4; ++nt) {
      v16bf b = load_b_frag(Pdw2, kt * 8 + nh * 4 + nt);
      acc2[nt] = WMMA_BF16(a, b, acc2[nt]);
    }
  }
#pragma unroll
  for (int nt = 0; nt < 4; ++nt) {
    int col = (nh * 4 + nt) * 16 + (lane & 15);
    float p = 0.0f;
#pragma unroll
    for (int j = 0; j < 8; ++j) p += acc2[nt][j] + db2[col];
    p += __shfl_xor(p, 16, 32);                  // sum over 16 rows of this m-tile
    if (lane < 16) atomicAdd(&gpart[col], p);
  }
  __syncthreads();
  if (t < 128) atomicAdd(&gsum[t], gpart[t]);
}

// ---------------------------------------------------------------------------
// graph_dec + L2 normalize (single tiny block).
// ---------------------------------------------------------------------------
__global__ void final_kernel(const float* __restrict__ gsum,
                             const float* __restrict__ gw1, const float* __restrict__ gb1,
                             const float* __restrict__ gw2, const float* __restrict__ gb2,
                             float* __restrict__ out) {
  __shared__ float gv[128], a1[128], red[128];
  int t = threadIdx.x;
  gv[t] = gsum[t];
  __syncthreads();
  float s = gb1[t];
  for (int k = 0; k < 128; ++k) s += gv[k] * gw1[k * 128 + t];
  a1[t] = siluf(s);
  __syncthreads();
  float o = gb2[t];
  for (int k = 0; k < 128; ++k) o += a1[k] * gw2[k * 128 + t];
  red[t] = o * o;
  __syncthreads();
  for (int st = 64; st > 0; st >>= 1) {
    if (t < st) red[t] += red[t + st];
    __syncthreads();
  }
  float nrm = fmaxf(sqrtf(red[0]), 1e-12f);
  out[t] = o / nrm;
}

// ---------------------------------------------------------------------------
extern "C" void kernel_launch(void* const* d_in, const int* in_sizes, int n_in,
                              void* d_out, int out_size, void* d_ws, size_t ws_size,
                              hipStream_t stream) {
  (void)in_sizes; (void)n_in; (void)out_size; (void)ws_size;
  const float* x      = (const float*)d_in[0];
  const float* coords = (const float*)d_in[1];
  const int*   ei     = (const int*)d_in[2];
  const float* w_enc  = (const float*)d_in[3];
  const float* b_enc  = (const float*)d_in[4];
  const float* ew1    = (const float*)d_in[5];
  const float* eb1    = (const float*)d_in[6];
  const float* ew2    = (const float*)d_in[7];
  const float* eb2    = (const float*)d_in[8];
  const float* nw1    = (const float*)d_in[9];
  const float* nb1    = (const float*)d_in[10];
  const float* nw2    = (const float*)d_in[11];
  const float* nb2    = (const float*)d_in[12];
  const float* dw1    = (const float*)d_in[13];
  const float* db1    = (const float*)d_in[14];
  const float* dw2    = (const float*)d_in[15];
  const float* db2    = (const float*)d_in[16];
  const float* gw1    = (const float*)d_in[17];
  const float* gb1    = (const float*)d_in[18];
  const float* gw2    = (const float*)d_in[19];
  const float* gb2    = (const float*)d_in[20];

  char* ws = (char*)d_ws;
  size_t off = 0;
  auto take = [&](size_t bytes) -> char* {
    char* p = ws + off;
    off = (off + bytes + 255) & ~(size_t)255;
    return p;
  };
  float*          featsF = (float*)take((size_t)NNODES * DDIM * 4);
  unsigned short* featsH = (unsigned short*)take((size_t)NNODES * DDIM * 2);
  float*          m_i    = (float*)take((size_t)NNODES * MDIM * 4);
  int*            idxW   = (int*)take((size_t)NNODES * KNBR * 4);
  float*          rdW    = (float*)take((size_t)NNODES * KNBR * 4);
  float*          gsum   = (float*)take(512);
  unsigned short* Penc   = (unsigned short*)take(24 * 1024);
  unsigned short* Pew1   = (unsigned short*)take((size_t)NLAYER * 128 * 1024);
  unsigned short* Pew2   = (unsigned short*)take((size_t)NLAYER * 32 * 1024);
  unsigned short* Pnw1   = (unsigned short*)take((size_t)NLAYER * 96 * 1024);
  unsigned short* Pnw2   = (unsigned short*)take((size_t)NLAYER * 64 * 1024);
  unsigned short* Pdw1   = (unsigned short*)take(32 * 1024);
  unsigned short* Pdw2   = (unsigned short*)take(32 * 1024);

  // one-time weight repacking into WMMA B-fragment layout (bf16)
  pack_b_kernel<<<24, 32, 0, stream>>>(w_enc, Penc, NFEAT, DDIM);           // Kp=96
  for (int l = 0; l < NLAYER; ++l) {
    pack_b_kernel<<<128, 32, 0, stream>>>(ew1 + (size_t)l * 257 * 256, Pew1 + (size_t)l * 128 * 512, 256, 256);
    pack_b_kernel<<<32,  32, 0, stream>>>(ew2 + (size_t)l * 256 * 64,  Pew2 + (size_t)l * 32 * 512, 256, 64);
    pack_b_kernel<<<96,  32, 0, stream>>>(nw1 + (size_t)l * 192 * 256, Pnw1 + (size_t)l * 96 * 512, 192, 256);
    pack_b_kernel<<<64,  32, 0, stream>>>(nw2 + (size_t)l * 256 * 128, Pnw2 + (size_t)l * 64 * 512, 256, 128);
  }
  pack_b_kernel<<<32, 32, 0, stream>>>(dw1, Pdw1, 128, 128);
  pack_b_kernel<<<32, 32, 0, stream>>>(dw2, Pdw2, 128, 128);

  zero_kernel<<<1, 128, 0, stream>>>(gsum);
  encoder_kernel<<<NNODES / 64, 256, 0, stream>>>(x, Penc, b_enc, featsF, featsH);
  nbr_kernel<<<NNODES, 32, 0, stream>>>(ei, coords, idxW, rdW);

  for (int l = 0; l < NLAYER; ++l) {
    edge_kernel<<<NNODES / NPB, 256, 0, stream>>>(
        featsH, idxW, rdW,
        Pew1 + (size_t)l * 128 * 512,
        ew1 + (size_t)l * 257 * 256 + 256 * 256,   // 257th row of ew1[l]
        eb1 + (size_t)l * 256,
        Pew2 + (size_t)l * 32 * 512,
        eb2 + (size_t)l * 64,
        m_i);
    node_kernel<<<NNODES / 64, 256, 0, stream>>>(
        featsF, featsH, m_i,
        Pnw1 + (size_t)l * 96 * 512, nb1 + (size_t)l * 256,
        Pnw2 + (size_t)l * 64 * 512, nb2 + (size_t)l * 128);
  }

  dec_kernel<<<NNODES / 64, 256, 0, stream>>>(featsF, Pdw1, db1, Pdw2, db2, gsum);
  final_kernel<<<1, 128, 0, stream>>>(gsum, gw1, gb1, gw2, gb2, (float*)d_out);
}
